// GraphConvNet_82729660055791
// MI455X (gfx1250) — matrix operand order repaired and checked by reading, not compiled, not run on hardware
//
#include <hip/hip_runtime.h>
#include <hip/hip_bf16.h>

// ---------------------------------------------------------------------------
// GCN (3 layers, C=128) for MI455X / gfx1250 (wave32, WMMA).
//  - BN folded into weights once per launch (weights -> f16, bias -> f32)
//  - edge aggregation: 1 wave per edge, float4/lane, global_atomic_add_f32,
//    edge indices fetched via SMEM (wave-uniform)
//  - per-vertex GEMMs: v_wmma_f32_16x16x32_f16, f32 accumulate
// ---------------------------------------------------------------------------

typedef __attribute__((ext_vector_type(16))) _Float16 v16h;
typedef __attribute__((ext_vector_type(8)))  _Float16 v8h;
typedef __attribute__((ext_vector_type(8)))  float    v8f;

constexpr int   Bsz    = 4;
constexpr int   Nv     = 20000;      // divisible by 16 -> no ragged tiles
constexpr int   Ne     = 320000;
constexpr int   Cc     = 128;
constexpr float EPS_BN = 1e-5f;

struct ParamPtrs {
  const float* W[6];
  const float* b[6];
  const float* gamma[6];
  const float* beta[6];
  const float* mean[6];
  const float* var[6];
};

// ---------------------------------------------------------------------------
// Fold BatchNorm into conv weights.  Block order: L0c1, L0c2, L1c1, L1c2,
// L2c1, L2c2.  Block 0 (3->128) stays f32 (VALU path); blocks 1..5 go to f16
// stored [cout][cin] so B-matrix loads are contiguous per lane.
// ---------------------------------------------------------------------------
__global__ __launch_bounds__(256) void fuse_params_kernel(
    ParamPtrs p, float* __restrict__ W0f, _Float16* __restrict__ Wf,
    float* __restrict__ bf)
{
  int blk = blockIdx.x;                       // 0..5
  int t   = threadIdx.x;
  int cin = (blk == 0) ? 3 : Cc;
  if (t < Cc) {
    float s = p.gamma[blk][t] * rsqrtf(p.var[blk][t] + EPS_BN);
    bf[blk * Cc + t] = (p.b[blk][t] - p.mean[blk][t]) * s + p.beta[blk][t];
  }
  for (int i = t; i < Cc * cin; i += 256) {
    int co = i / cin;
    float s = p.gamma[blk][co] * rsqrtf(p.var[blk][co] + EPS_BN);
    float w = p.W[blk][i] * s;
    if (blk == 0) W0f[i] = w;
    else          Wf[(size_t)(blk - 1) * Cc * Cc + i] = (_Float16)w;
  }
}

__global__ void zero_kernel(float* __restrict__ p, size_t n)
{
  size_t i = (size_t)blockIdx.x * blockDim.x + threadIdx.x;
  if (i < n) p[i] = 0.0f;
}

// ---------------------------------------------------------------------------
// Layer 0 aggregation: scatter-add (x[to]-x[fr]) over 3 coords, 1 thread/edge.
// ---------------------------------------------------------------------------
__global__ __launch_bounds__(256) void aggregate0_kernel(
    const float* __restrict__ verts, const int* __restrict__ edges,
    float* __restrict__ aggr3)
{
  size_t idx = (size_t)blockIdx.x * 256 + threadIdx.x;
  if (idx >= (size_t)Bsz * Ne) return;
  int b = (int)(idx / Ne), e = (int)(idx % Ne);
  const int* eb = edges + (size_t)b * 2 * Ne;
  int fr = eb[e], to = eb[Ne + e];
  const float* vb = verts + (size_t)b * Nv * 3;
  float*       ab = aggr3 + (size_t)b * Nv * 3;
#pragma unroll
  for (int c = 0; c < 3; ++c)
    atomicAdd(&ab[fr * 3 + c], vb[to * 3 + c] - vb[fr * 3 + c]);
}

// ---------------------------------------------------------------------------
// C=128 aggregation: one wave per edge, float4 per lane (512B gather + 512B
// atomic RMW per edge -> the dominant, L2-resident traffic).  The edge id is
// wave-uniform; readfirstlane moves it to SGPRs so fr/to come in via SMEM.
// ---------------------------------------------------------------------------
__global__ __launch_bounds__(256) void aggregateC_kernel(
    const float* __restrict__ x, const int* __restrict__ edges,
    float* __restrict__ aggr)
{
  int gw = __builtin_amdgcn_readfirstlane(
      (int)(((size_t)blockIdx.x * 256 + threadIdx.x) >> 5));
  if (gw >= Bsz * Ne) return;                 // wave-uniform branch
  int lane = threadIdx.x & 31;
  int b = gw / Ne, e = gw % Ne;
  const int* eb = edges + (size_t)b * 2 * Ne;
  int fr = eb[e];                             // uniform -> s_load
  int to = eb[Ne + e];                        // uniform -> s_load
  const float4 v = ((const float4*)(x + ((size_t)b * Nv + to) * Cc))[lane];
  float* dst = aggr + ((size_t)b * Nv + fr) * Cc + lane * 4;
  atomicAdd(dst + 0, v.x); atomicAdd(dst + 1, v.y);
  atomicAdd(dst + 2, v.z); atomicAdd(dst + 3, v.w);
}

// ---------------------------------------------------------------------------
// One 16x16 output tile of [16 x 128] x [128 x 128]^T via 4x wmma K=32.
// A in LDS, row-major f16 [16][128]; lane layout per ISA 7.12.2:
//   lanes 0-15 hold K {0..7,16..23}+Kc, lanes 16-31 hold K {8..15,24..31}+Kc.
// W stored [cout][cin] f16 -> each lane's 16 B-values are two 16-byte runs, same K as A.
// ---------------------------------------------------------------------------
__device__ inline v8f wmma_tile_16x128(const _Float16* ldsA,
                                       const _Float16* __restrict__ W,
                                       int wave, int lane)
{
  int m  = lane & 15;
  int hi = lane >> 4;
  int ko = hi * 8;      // A: K sub-offset for this lane half
  int kb = hi * 8;      // B: K base for this lane half
  int col = wave * 16 + m;
  v8f acc = {};
#pragma unroll
  for (int kt = 0; kt < 4; ++kt) {
    int Kc = kt * 32;
    v8h alo = *(const v8h*)(ldsA + m * Cc + Kc + ko);
    v8h ahi = *(const v8h*)(ldsA + m * Cc + Kc + 16 + ko);
    v8h blo = *(const v8h*)(W + (size_t)col * Cc + Kc + kb);
    v8h bhi = *(const v8h*)(W + (size_t)col * Cc + Kc + kb + 16);
    v16h a, bb;
#pragma unroll
    for (int j = 0; j < 8; ++j) {
      a[j] = alo[j]; a[8 + j] = ahi[j];
      bb[j] = blo[j]; bb[8 + j] = bhi[j];
    }
    acc = __builtin_amdgcn_wmma_f32_16x16x32_f16(
        false, a, false, bb, (short)0, acc, false, false);
  }
  return acc;
}

// ---------------------------------------------------------------------------
// Layer 0: x = (aggr3 / d) -> conv_bn(3->128) via VALU -> conv_bn(128->128)
// via WMMA.  One 16-vertex tile per 256-thread block (8 waves = 8 col-tiles).
// ---------------------------------------------------------------------------
__global__ __launch_bounds__(256) void layer0_kernel(
    const float* __restrict__ aggr3, const float* __restrict__ dofs,
    const float* __restrict__ W0f, const float* __restrict__ bf,
    const _Float16* __restrict__ Wf, float* __restrict__ xA)
{
  __shared__ float    in0[16][3];
  __shared__ _Float16 ldsH[16 * Cc];
  size_t row0 = (size_t)blockIdx.x * 16;
  int t = threadIdx.x;
  if (t < 48) {
    int r = t / 3, c = t % 3;
    size_t g = row0 + r;
    in0[r][c] = aggr3[g * 3 + c] * (1.0f / dofs[g]);
  }
  __syncthreads();
  const float* b1 = bf + 0 * Cc;            // L0c1 bias
  for (int i = t; i < 16 * Cc; i += 256) {
    int r = i >> 7, co = i & 127;
    float h = b1[co] + in0[r][0] * W0f[co * 3 + 0]
                     + in0[r][1] * W0f[co * 3 + 1]
                     + in0[r][2] * W0f[co * 3 + 2];
    ldsH[r * Cc + co] = (_Float16)h;
  }
  __syncthreads();
  int wave = t >> 5, lane = t & 31;
  v8f acc = wmma_tile_16x128(ldsH, Wf /* L0c2 */, wave, lane);
  int m = lane & 15, cm0 = (lane < 16) ? 0 : 8;
  float bias = bf[1 * Cc + wave * 16 + m];  // L0c2 bias
#pragma unroll
  for (int j = 0; j < 8; ++j)
    xA[(row0 + cm0 + j) * Cc + wave * 16 + m] = acc[j] + bias;
}

// ---------------------------------------------------------------------------
// Layers 1/2: x' = x + aggr/d -> two 128x128 GEMMs (WMMA) -> optional ReLU.
// In-place (xout may alias xin): each block reads its 16 rows fully into LDS
// before writing them.  Staging is fully vectorized: each thread owns 8
// contiguous channels of one row (2x b128 from x, 2x b128 from aggr, 1x b128
// LDS store of 8 halfs) with one 1/d reciprocal per row.
// ---------------------------------------------------------------------------
__global__ __launch_bounds__(256) void layerC_kernel(
    const float* __restrict__ xin, const float* __restrict__ aggr,
    const float* __restrict__ dofs,
    const _Float16* __restrict__ W1, const float* __restrict__ b1,
    const _Float16* __restrict__ W2, const float* __restrict__ b2,
    float* xout, int relu)
{
  __shared__ _Float16 ldsA[16 * Cc];
  __shared__ _Float16 ldsH[16 * Cc];
  size_t row0 = (size_t)blockIdx.x * 16;
  int t = threadIdx.x;

  {
    int r  = t >> 4;            // 0..15
    int c0 = (t & 15) * 8;      // 0..120
    size_t g = row0 + r;
    float invd = 1.0f / dofs[g];
    const float4* xp = (const float4*)(xin  + g * Cc + c0);
    const float4* ap = (const float4*)(aggr + g * Cc + c0);
    float4 x0 = xp[0], x1 = xp[1];
    float4 a0 = ap[0], a1 = ap[1];
    v8h hv;
    hv[0] = (_Float16)(x0.x + a0.x * invd);
    hv[1] = (_Float16)(x0.y + a0.y * invd);
    hv[2] = (_Float16)(x0.z + a0.z * invd);
    hv[3] = (_Float16)(x0.w + a0.w * invd);
    hv[4] = (_Float16)(x1.x + a1.x * invd);
    hv[5] = (_Float16)(x1.y + a1.y * invd);
    hv[6] = (_Float16)(x1.z + a1.z * invd);
    hv[7] = (_Float16)(x1.w + a1.w * invd);
    *(v8h*)(ldsA + r * Cc + c0) = hv;
  }
  __syncthreads();

  int wave = t >> 5, lane = t & 31;
  int m = lane & 15, cm0 = (lane < 16) ? 0 : 8;

  v8f h = wmma_tile_16x128(ldsA, W1, wave, lane);
  float bias1 = b1[wave * 16 + m];
#pragma unroll
  for (int j = 0; j < 8; ++j)
    ldsH[(cm0 + j) * Cc + wave * 16 + m] = (_Float16)(h[j] + bias1);
  __syncthreads();

  v8f o = wmma_tile_16x128(ldsH, W2, wave, lane);
  float bias2 = b2[wave * 16 + m];
#pragma unroll
  for (int j = 0; j < 8; ++j) {
    float v = o[j] + bias2;
    if (relu) v = fmaxf(v, 0.0f);
    xout[(row0 + cm0 + j) * Cc + wave * 16 + m] = v;
  }
}

// ---------------------------------------------------------------------------
extern "C" void kernel_launch(void* const* d_in, const int* in_sizes, int n_in,
                              void* d_out, int out_size, void* d_ws,
                              size_t ws_size, hipStream_t stream)
{
  // Identify the three big inputs by exact flat size; params = 36 leaves
  // starting at the first 128x3 weight, order per setup_inputs():
  // per layer: c1{W,b,gamma,beta,mean,var}, c2{...}.
  const float* vertices = nullptr;
  const int*   edges    = nullptr;
  const float* dofs     = nullptr;
  int pstart = -1;
  for (int i = 0; i < n_in; ++i) {
    if (in_sizes[i] == Bsz * Nv * 3 && !vertices) vertices = (const float*)d_in[i];
    else if (in_sizes[i] == Bsz * 2 * Ne && !edges) edges = (const int*)d_in[i];
    else if (in_sizes[i] == Bsz * Nv && !dofs) dofs = (const float*)d_in[i];
    else if (in_sizes[i] == Cc * 3 && pstart < 0) pstart = i;
  }
  if (!vertices || !edges || !dofs || pstart < 0) return;

  ParamPtrs pp;
  for (int blk = 0; blk < 6; ++blk) {
    int base = pstart + blk * 6;
    pp.W[blk]     = (const float*)d_in[base + 0];
    pp.b[blk]     = (const float*)d_in[base + 1];
    pp.gamma[blk] = (const float*)d_in[base + 2];
    pp.beta[blk]  = (const float*)d_in[base + 3];
    pp.mean[blk]  = (const float*)d_in[base + 4];
    pp.var[blk]   = (const float*)d_in[base + 5];
  }

  // Workspace carve-out (256B aligned).
  char* ws = (char*)d_ws;
  size_t off = 0;
  auto carve = [&](size_t bytes) -> char* {
    size_t o = (off + 255) & ~(size_t)255;
    off = o + bytes;
    return ws + o;
  };
  float*    W0f   = (float*)   carve(Cc * 3 * sizeof(float));
  _Float16* Wf    = (_Float16*)carve((size_t)5 * Cc * Cc * sizeof(_Float16));
  float*    bf    = (float*)   carve(6 * Cc * sizeof(float));
  float*    aggr3 = (float*)   carve((size_t)Bsz * Nv * 3 * sizeof(float));
  float*    aggrC = (float*)   carve((size_t)Bsz * Nv * Cc * sizeof(float));
  float*    xA    = (float*)   carve((size_t)Bsz * Nv * Cc * sizeof(float));
  if (off > ws_size) return;

  const _Float16* W_L0c2 = Wf + (size_t)0 * Cc * Cc;
  const _Float16* W_L1c1 = Wf + (size_t)1 * Cc * Cc;
  const _Float16* W_L1c2 = Wf + (size_t)2 * Cc * Cc;
  const _Float16* W_L2c1 = Wf + (size_t)3 * Cc * Cc;
  const _Float16* W_L2c2 = Wf + (size_t)4 * Cc * Cc;

  const int tiles     = Bsz * Nv / 16;                 // 5000
  const int edgeThr   = (Bsz * Ne + 255) / 256;        // 1 thread/edge
  const int edgeWaveB = (Bsz * Ne * 32 + 255) / 256;   // 1 wave/edge
  const size_t nAgg3  = (size_t)Bsz * Nv * 3;
  const size_t nAggC  = (size_t)Bsz * Nv * Cc;

  fuse_params_kernel<<<6, 256, 0, stream>>>(pp, W0f, Wf, bf);

  // ---- Layer 0 ----
  zero_kernel<<<(int)((nAgg3 + 255) / 256), 256, 0, stream>>>(aggr3, nAgg3);
  aggregate0_kernel<<<edgeThr, 256, 0, stream>>>(vertices, edges, aggr3);
  layer0_kernel<<<tiles, 256, 0, stream>>>(aggr3, dofs, W0f, bf, W_L0c2, xA);

  // ---- Layer 1 (ReLU) ----
  zero_kernel<<<(int)((nAggC + 255) / 256), 256, 0, stream>>>(aggrC, nAggC);
  aggregateC_kernel<<<edgeWaveB, 256, 0, stream>>>(xA, edges, aggrC);
  layerC_kernel<<<tiles, 256, 0, stream>>>(xA, aggrC, dofs,
                                           W_L1c1, bf + 2 * Cc,
                                           W_L1c2, bf + 3 * Cc, xA, 1);

  // ---- Layer 2 ----
  zero_kernel<<<(int)((nAggC + 255) / 256), 256, 0, stream>>>(aggrC, nAggC);
  aggregateC_kernel<<<edgeWaveB, 256, 0, stream>>>(xA, edges, aggrC);
  layerC_kernel<<<tiles, 256, 0, stream>>>(xA, aggrC, dofs,
                                           W_L2c1, bf + 4 * Cc,
                                           W_L2c2, bf + 5 * Cc,
                                           (float*)d_out, 0);
}
